// Seq2SeqBiLSTM_77171972374575
// MI455X (gfx1250) — matrix operand order repaired and checked by reading, not compiled, not run on hardware
//
#include <hip/hip_runtime.h>
#include <hip/hip_bf16.h>
#include <math.h>

#define NEG_INF_F (-1000000000.0f)

typedef __attribute__((ext_vector_type(16))) __bf16 v16bf;
typedef __attribute__((ext_vector_type(8)))  __bf16 v8bf;
typedef __attribute__((ext_vector_type(8)))  float  v8f;

// ---- vectorized fragment loads (contiguous per-lane per CDNA5 ISA 7.12.2) ----
// A (16-bit, 16x32): lane holds K = {k0+8h .. +7} and {k0+16+8h .. +7}  (h = lane>=16)
__device__ __forceinline__ v16bf ld_afrag(const __bf16* __restrict__ p) {
  v8bf lo = *(const v8bf*)(p);
  v8bf hi = *(const v8bf*)(p + 16);
  v16bf r;
#pragma unroll
  for (int e = 0; e < 8; ++e) { r[e] = lo[e]; r[e + 8] = hi[e]; }
  return r;
}
// B (16-bit, 32x16): lane holds K = {k0+8h .. +7} and {k0+16+8h .. +7} of column (n0+lane%16)
__device__ __forceinline__ v16bf ld_bfrag(const __bf16* __restrict__ p) {
  v8bf lo = *(const v8bf*)(p);
  v8bf hi = *(const v8bf*)(p + 16);
  v16bf r;
#pragma unroll
  for (int e = 0; e < 8; ++e) { r[e] = lo[e]; r[e + 8] = hi[e]; }
  return r;
}

// One 16x(16*NW) output block per wave; A-fragment reused across NW B-tiles.
// Software-pipelined double buffering: loads for k0+32 are issued before the
// WMMAs for k0, so v_wmma overlaps with in-flight global_load_b128s.
// A row-major activations A[row*lda + k]; W row-major weights (N x K) -> X @ W^T.
template <int NW>
__device__ __forceinline__ void wmma_kloop(const __bf16* __restrict__ A, int lda,
                                           const __bf16* __restrict__ W, int ldw,
                                           int m0, int n0, int K, v8f* acc,
                                           int prefetch = 0) {
  const int lane = threadIdx.x & 31;
  const int half = lane >> 4;
  const int lmod = lane & 15;
  const __bf16* ap = A + (m0 + lmod) * lda + 8 * half;
  const __bf16* wp = W + lmod * ldw + 8 * half;
  v16bf a = ld_afrag(ap);
  v16bf b[NW];
#pragma unroll
  for (int w = 0; w < NW; ++w) b[w] = ld_bfrag(wp + (n0 + 16 * w) * ldw);
  for (int k0 = 0; k0 < K; k0 += 32) {
    const int kn = (k0 + 32 < K) ? (k0 + 32) : k0;   // branchless pipeline tail
    v16bf an = ld_afrag(ap + kn);
    v16bf bn[NW];
#pragma unroll
    for (int w = 0; w < NW; ++w) {
      const __bf16* bp = wp + (n0 + 16 * w) * ldw + kn;
      bn[w] = ld_bfrag(bp);
      if (prefetch && (k0 + 64) < K) __builtin_prefetch(bp + 32, 0, 1);
    }
#pragma unroll
    for (int w = 0; w < NW; ++w)
      acc[w] = __builtin_amdgcn_wmma_f32_16x16x32_bf16(false, a, false, b[w],
                                                       (short)0, acc[w], false, false);
    a = an;
#pragma unroll
    for (int w = 0; w < NW; ++w) b[w] = bn[w];
  }
}

// ---------------------------------------------------------------- utilities
__global__ void k_f2bf(const float* __restrict__ in, __bf16* __restrict__ out, int n) {
  int i = blockIdx.x * blockDim.x + threadIdx.x;
  if (i < n) out[i] = (__bf16)in[i];
}

// gather src_emb rows per token -> bf16 (2048 x 768)
__global__ void k_gather_src(const int* __restrict__ xs, const float* __restrict__ emb,
                             __bf16* __restrict__ out) {
  int i = blockIdx.x * blockDim.x + threadIdx.x;
  if (i >= 2048 * 768) return;
  int row = i / 768;
  int p   = i - row * 768;
  out[i] = (__bf16)emb[(long)xs[row] * 768 + p];
}

// ------------------------------------------------------------- generic GEMM
// out[m][n] = sum_k A_bf[m][k] * W_bf[n][k] + bias[n]; 1x4 tile blocking per wave.
// Requires M%16==0, N%64==0, K%32==0.
__global__ void k_gemm(const __bf16* __restrict__ A, int lda,
                       const __bf16* __restrict__ W, int ldw,
                       const float* __restrict__ bias,
                       float* __restrict__ outF, __bf16* __restrict__ outB,
                       int ldo, int M, int N, int K, int relu) {
  const int wave = threadIdx.x >> 5;
  const int nJ = N >> 6;                          // groups of 4 n-tiles
  const long jobs = (long)(M >> 4) * nJ;
  const long job = (long)blockIdx.x * (blockDim.x >> 5) + wave;
  if (job >= jobs) return;                        // wave-uniform exit
  const int m0 = (int)(job / nJ) << 4;
  const int n0 = (int)(job % nJ) << 6;
  const int lane = threadIdx.x & 31;
  const int half = lane >> 4;
  const int lmod = lane & 15;
  v8f acc[4];
#pragma unroll
  for (int w = 0; w < 4; ++w) {
    const float bv = bias ? bias[n0 + 16 * w + lmod] : 0.0f;
#pragma unroll
    for (int r = 0; r < 8; ++r) acc[w][r] = bv;
  }
  wmma_kloop<4>(A, lda, W, ldw, m0, n0, K, acc, /*prefetch=*/1);
#pragma unroll
  for (int w = 0; w < 4; ++w) {
#pragma unroll
    for (int r = 0; r < 8; ++r) {
      float v = acc[w][r];
      if (relu) v = v > 0.0f ? v : 0.0f;
      const long idx = (long)(m0 + r + 8 * half) * ldo + (n0 + 16 * w + lmod);
      if (outF) outF[idx] = v;
      if (outB) outB[idx] = (__bf16)v;
    }
  }
}

// ----------------------------------------------------- encoder BiLSTM (2 blocks)
// gi_* = x @ Wih^T + b (precomputed, (2048,1024)).  Per step: gates = gi + h@Whh^T,
// masked LSTM update. dir 0 = forward, dir 1 = reversed-input backward pass.
__global__ void k_enc_lstm(const float* __restrict__ gi_f, const float* __restrict__ gi_b,
                           const __bf16* __restrict__ whhf, const __bf16* __restrict__ whhb,
                           const int* __restrict__ lens,
                           float* __restrict__ out_f, float* __restrict__ out_r,
                           float* __restrict__ dec_h, float* __restrict__ dec_c,
                           float* __restrict__ h_st, float* __restrict__ c_st,
                           __bf16* __restrict__ hbf_st, float* __restrict__ gates_st) {
  const int dir = blockIdx.x;
  const float* gi   = dir ? gi_b : gi_f;
  const __bf16* whh = dir ? whhb : whhf;
  float* outp  = dir ? out_r : out_f;
  float* h     = h_st     + dir * 32 * 256;
  float* c     = c_st     + dir * 32 * 256;
  __bf16* hbf  = hbf_st   + dir * 32 * 256;
  float* gates = gates_st + dir * 32 * 1024;
  const int tid  = threadIdx.x;
  const int wave = tid >> 5;
  const int lane = tid & 31;
  const int half = lane >> 4;
  const int lmod = lane & 15;

  for (int i = tid; i < 32 * 256; i += 1024) { h[i] = 0.f; c[i] = 0.f; hbf[i] = (__bf16)0.f; }
  __syncthreads();

  for (int t = 0; t < 64; ++t) {
    // gates = gi(row for this step) + h @ Whh^T; 2x64 tiles -> 32 jobs of 1x4, one per wave
    {
      const int m0 = (wave >> 4) << 4;       // 0 or 16
      const int n0 = (wave & 15) << 6;
      v8f acc[4];
#pragma unroll
      for (int r = 0; r < 8; ++r) {
        const int b = m0 + r + 8 * half;
        int sidx = t;
        if (dir) {
          sidx = lens[b] - 1 - t;
          if (sidx < 0) sidx = 0;
          if (sidx > 63) sidx = 63;
        }
        const float* grow = gi + (b * 64 + sidx) * 1024;
#pragma unroll
        for (int w = 0; w < 4; ++w) acc[w][r] = grow[n0 + 16 * w + lmod];
      }
      wmma_kloop<4>(hbf, 256, whh, 256, m0, n0, 256, acc);
#pragma unroll
      for (int w = 0; w < 4; ++w)
#pragma unroll
        for (int r = 0; r < 8; ++r)
          gates[(m0 + r + 8 * half) * 1024 + (n0 + 16 * w + lmod)] = acc[w][r];
    }
    __syncthreads();
    // elementwise masked LSTM update
    for (int i = tid; i < 32 * 256; i += 1024) {
      const int b = i >> 8;
      const int hh = i & 255;
      const float giv = gates[b * 1024 + hh];
      const float gfv = gates[b * 1024 + 256 + hh];
      const float ggv = gates[b * 1024 + 512 + hh];
      const float gov = gates[b * 1024 + 768 + hh];
      const float si = 1.f / (1.f + expf(-giv));
      const float sf = 1.f / (1.f + expf(-gfv));
      const float so = 1.f / (1.f + expf(-gov));
      const float cn = sf * c[i] + si * tanhf(ggv);
      const float hn = so * tanhf(cn);
      const bool v = t < lens[b];
      const float hN = v ? hn : h[i];
      const float cN = v ? cn : c[i];
      h[i] = hN; c[i] = cN; hbf[i] = (__bf16)hN;
      outp[(b * 64 + t) * 256 + hh] = v ? hn : 0.f;
    }
    __syncthreads();
  }
  for (int i = tid; i < 32 * 256; i += 1024) {
    const int b = i >> 8;
    const int hh = i & 255;
    dec_h[b * 512 + dir * 256 + hh] = h[i];
    dec_c[b * 512 + dir * 256 + hh] = c[i];
  }
}

// enc[b,t] = [out_f[b,t], out_r[b, len-1-t] (masked)]
__global__ void k_assemble_enc(const float* __restrict__ out_f, const float* __restrict__ out_r,
                               const int* __restrict__ lens,
                               float* __restrict__ encF, __bf16* __restrict__ encB) {
  int i = blockIdx.x * blockDim.x + threadIdx.x;
  if (i >= 2048 * 512) return;
  const int j = i & 511;
  const int row = i >> 9;       // b*64 + t
  const int b = row >> 6;
  const int t = row & 63;
  float v;
  if (j < 256) {
    v = out_f[row * 256 + j];
  } else {
    const int L = lens[b];
    v = (t < L) ? out_r[(b * 64 + (L - 1 - t)) * 256 + (j - 256)] : 0.f;
  }
  encF[i] = v;
  encB[i] = (__bf16)v;
}

// ----------------------------------------------- decoder (1 block, persistent)
__global__ void k_decoder(const int* __restrict__ ys, const float* __restrict__ tgt_emb,
                          const int* __restrict__ lens,
                          const float* __restrict__ enc, const float* __restrict__ enc_proj,
                          const __bf16* __restrict__ wattn, const float* __restrict__ b_attn,
                          const float* __restrict__ v_attn,
                          const __bf16* __restrict__ wcomb, const float* __restrict__ b_comb,
                          const __bf16* __restrict__ wihd, const __bf16* __restrict__ whhd,
                          const float* __restrict__ b_d,
                          float* __restrict__ h, float* __restrict__ c,
                          __bf16* __restrict__ hid_bf, __bf16* __restrict__ xc_bf,
                          __bf16* __restrict__ xh_bf,
                          float* __restrict__ a_h, float* __restrict__ gates,
                          __bf16* __restrict__ h_all_bf) {
  __shared__ float sc[32 * 64];
  const int tid  = threadIdx.x;
  const int wave = tid >> 5;
  const int lane = tid & 31;
  const int half = lane >> 4;
  const int lmod = lane & 15;

  for (int t = 0; t < 64; ++t) {
    // P1: stage bf16 inputs: hid=[h,c], xh[:,512:]=h, xc[:,:512]=emb(tok)
    for (int i = tid; i < 32 * 512; i += 1024) {
      const int b = i >> 9;
      const int e = i & 511;
      const float hv = h[i];
      const float cv = c[i];
      hid_bf[b * 1024 + e]       = (__bf16)hv;
      hid_bf[b * 1024 + 512 + e] = (__bf16)cv;
      xh_bf[b * 1024 + 512 + e]  = (__bf16)hv;
      const int tok = (t == 0) ? 1 : ys[b * 65 + t];   // START_IDX = 1
      xc_bf[b * 1024 + e] = (__bf16)tgt_emb[tok * 512 + e];
    }
    __syncthreads();
    // P2: a_h = hid @ W_h^T + b_attn (W_h = W_attn[:, :1024]); 2x32 tiles -> 32 jobs of 1x2
    {
      const int m0 = (wave >> 4) << 4;
      const int n0 = (wave & 15) << 5;
      v8f acc[2];
#pragma unroll
      for (int w = 0; w < 2; ++w) {
        const float bv = b_attn[n0 + 16 * w + lmod];
#pragma unroll
        for (int r = 0; r < 8; ++r) acc[w][r] = bv;
      }
      wmma_kloop<2>(hid_bf, 1024, wattn, 1536, m0, n0, 1024, acc);
#pragma unroll
      for (int w = 0; w < 2; ++w)
#pragma unroll
        for (int r = 0; r < 8; ++r)
          a_h[(m0 + r + 8 * half) * 512 + (n0 + 16 * w + lmod)] = acc[w][r];
    }
    __syncthreads();
    // P3: scores[b,s] = v_attn . tanh(enc_proj[b,s] + a_h[b]) (masked)
    for (int idx = tid; idx < 2048; idx += 1024) {
      const int b = idx >> 6;
      const int s = idx & 63;
      float sv = NEG_INF_F;
      if (s < lens[b]) {
        float acc = 0.f;
        const float* ep = enc_proj + (b * 64 + s) * 512;
        const float* ah = a_h + b * 512;
        for (int e = 0; e < 512; ++e) acc += tanhf(ep[e] + ah[e]) * v_attn[e];
        sv = acc;
      }
      sc[idx] = sv;
    }
    __syncthreads();
    // P4: softmax over s per batch row
    if (tid < 32) {
      float m = -1e30f;
      for (int s = 0; s < 64; ++s) m = fmaxf(m, sc[tid * 64 + s]);
      float sum = 0.f;
      for (int s = 0; s < 64; ++s) { float e = expf(sc[tid * 64 + s] - m); sc[tid * 64 + s] = e; sum += e; }
      const float inv = 1.f / sum;
      for (int s = 0; s < 64; ++s) sc[tid * 64 + s] *= inv;
    }
    __syncthreads();
    // P5: ctx -> xc[:, 512:]
    for (int i = tid; i < 32 * 512; i += 1024) {
      const int b = i >> 9;
      const int e = i & 511;
      float acc = 0.f;
      for (int s = 0; s < 64; ++s) acc += sc[b * 64 + s] * enc[(b * 64 + s) * 512 + e];
      xc_bf[b * 1024 + 512 + e] = (__bf16)acc;
    }
    __syncthreads();
    // P6: xi = relu(xc @ Wcomb^T + b_comb) -> xh[:, :512]; 32 jobs of 1x2
    {
      const int m0 = (wave >> 4) << 4;
      const int n0 = (wave & 15) << 5;
      v8f acc[2];
#pragma unroll
      for (int w = 0; w < 2; ++w) {
        const float bv = b_comb[n0 + 16 * w + lmod];
#pragma unroll
        for (int r = 0; r < 8; ++r) acc[w][r] = bv;
      }
      wmma_kloop<2>(xc_bf, 1024, wcomb, 1024, m0, n0, 1024, acc);
#pragma unroll
      for (int w = 0; w < 2; ++w)
#pragma unroll
        for (int r = 0; r < 8; ++r) {
          float v = acc[w][r];
          v = v > 0.f ? v : 0.f;
          xh_bf[(m0 + r + 8 * half) * 1024 + (n0 + 16 * w + lmod)] = (__bf16)v;
        }
    }
    __syncthreads();
    // P7: gates = [xi, h] @ [Wih_d ; Whh_d]^T + b_d; 2x128 tiles -> 64 jobs of 1x4
    for (int job = wave; job < 64; job += 32) {
      const int m0 = (job >> 5) << 4;
      const int n0 = (job & 31) << 6;
      v8f acc[4];
#pragma unroll
      for (int w = 0; w < 4; ++w) {
        const float bv = b_d[n0 + 16 * w + lmod];
#pragma unroll
        for (int r = 0; r < 8; ++r) acc[w][r] = bv;
      }
      wmma_kloop<4>(xh_bf,       1024, wihd, 512, m0, n0, 512, acc);
      wmma_kloop<4>(xh_bf + 512, 1024, whhd, 512, m0, n0, 512, acc);
#pragma unroll
      for (int w = 0; w < 4; ++w)
#pragma unroll
        for (int r = 0; r < 8; ++r)
          gates[(m0 + r + 8 * half) * 2048 + (n0 + 16 * w + lmod)] = acc[w][r];
    }
    __syncthreads();
    // P8: LSTM update, stash h_t as bf16 in (b,t) order for the batched W_out GEMM
    for (int i = tid; i < 32 * 512; i += 1024) {
      const int b = i >> 9;
      const int e = i & 511;
      const float giv = gates[b * 2048 + e];
      const float gfv = gates[b * 2048 + 512 + e];
      const float ggv = gates[b * 2048 + 1024 + e];
      const float gov = gates[b * 2048 + 1536 + e];
      const float si = 1.f / (1.f + expf(-giv));
      const float sf = 1.f / (1.f + expf(-gfv));
      const float so = 1.f / (1.f + expf(-gov));
      const float cn = sf * c[i] + si * tanhf(ggv);
      const float hn = so * tanhf(cn);
      c[i] = cn;
      h[i] = hn;
      h_all_bf[(b * 64 + t) * 512 + e] = (__bf16)hn;
    }
    __syncthreads();
  }
}

// in-place log_softmax over V=32000 per row, 1 block per row
__global__ void k_logsoftmax(float* __restrict__ out) {
  __shared__ float red[256];
  float* x = out + (long)blockIdx.x * 32000;
  const int tid = threadIdx.x;
  float m = -1e30f;
  for (int v = tid; v < 32000; v += 256) m = fmaxf(m, x[v]);
  red[tid] = m;
  __syncthreads();
  for (int s = 128; s > 0; s >>= 1) { if (tid < s) red[tid] = fmaxf(red[tid], red[tid + s]); __syncthreads(); }
  m = red[0];
  __syncthreads();
  float sum = 0.f;
  for (int v = tid; v < 32000; v += 256) sum += expf(x[v] - m);
  red[tid] = sum;
  __syncthreads();
  for (int s = 128; s > 0; s >>= 1) { if (tid < s) red[tid] += red[tid + s]; __syncthreads(); }
  const float ls = logf(red[0]);
  __syncthreads();
  for (int v = tid; v < 32000; v += 256) x[v] = x[v] - m - ls;
}

// =============================================================== launch
extern "C" void kernel_launch(void* const* d_in, const int* in_sizes, int n_in,
                              void* d_out, int out_size, void* d_ws, size_t ws_size,
                              hipStream_t stream) {
  (void)in_sizes; (void)n_in; (void)out_size; (void)ws_size;
  const int*   xs      = (const int*)d_in[0];
  const int*   x_lens  = (const int*)d_in[1];
  const int*   ys      = (const int*)d_in[2];
  const float* src_emb = (const float*)d_in[3];
  const float* W_tr    = (const float*)d_in[4];
  const float* b_tr    = (const float*)d_in[5];
  const float* Wih_f   = (const float*)d_in[6];
  const float* Whh_f   = (const float*)d_in[7];
  const float* b_f     = (const float*)d_in[8];
  const float* Wih_b   = (const float*)d_in[9];
  const float* Whh_b   = (const float*)d_in[10];
  const float* b_b     = (const float*)d_in[11];
  const float* W_attn  = (const float*)d_in[12];
  const float* b_attn  = (const float*)d_in[13];
  const float* v_attn  = (const float*)d_in[14];
  const float* tgt_emb = (const float*)d_in[15];
  const float* W_comb  = (const float*)d_in[16];
  const float* b_comb  = (const float*)d_in[17];
  const float* Wih_d   = (const float*)d_in[18];
  const float* Whh_d   = (const float*)d_in[19];
  const float* b_d     = (const float*)d_in[20];
  const float* W_out   = (const float*)d_in[21];
  const float* b_out   = (const float*)d_in[22];
  float* out = (float*)d_out;

  char* p = (char*)d_ws;
  auto alloc = [&](size_t bytes) -> void* {
    void* r = (void*)p;
    p += (bytes + 255) & ~(size_t)255;
    return r;
  };
  // bf16 weights
  __bf16* wtr_bf   = (__bf16*)alloc((size_t)512 * 768 * 2);
  __bf16* wihf_bf  = (__bf16*)alloc((size_t)1024 * 512 * 2);
  __bf16* wihb_bf  = (__bf16*)alloc((size_t)1024 * 512 * 2);
  __bf16* whhf_bf  = (__bf16*)alloc((size_t)1024 * 256 * 2);
  __bf16* whhb_bf  = (__bf16*)alloc((size_t)1024 * 256 * 2);
  __bf16* wattn_bf = (__bf16*)alloc((size_t)512 * 1536 * 2);
  __bf16* wcomb_bf = (__bf16*)alloc((size_t)512 * 1024 * 2);
  __bf16* wihd_bf  = (__bf16*)alloc((size_t)2048 * 512 * 2);
  __bf16* whhd_bf  = (__bf16*)alloc((size_t)2048 * 512 * 2);
  __bf16* wout_bf  = (__bf16*)alloc((size_t)32000 * 512 * 2);
  // activations
  __bf16* xemb_bf  = (__bf16*)alloc((size_t)2048 * 768 * 2);
  __bf16* x_bf     = (__bf16*)alloc((size_t)2048 * 512 * 2);
  float*  gi_f     = (float*)alloc((size_t)2048 * 1024 * 4);
  float*  gi_b     = (float*)alloc((size_t)2048 * 1024 * 4);
  float*  out_f    = (float*)alloc((size_t)2048 * 256 * 4);
  float*  out_r    = (float*)alloc((size_t)2048 * 256 * 4);
  float*  enc_f    = (float*)alloc((size_t)2048 * 512 * 4);
  __bf16* enc_bf   = (__bf16*)alloc((size_t)2048 * 512 * 2);
  float*  enc_proj = (float*)alloc((size_t)2048 * 512 * 4);
  float*  dec_h    = (float*)alloc((size_t)32 * 512 * 4);
  float*  dec_c    = (float*)alloc((size_t)32 * 512 * 4);
  float*  h_enc    = (float*)alloc((size_t)2 * 32 * 256 * 4);
  float*  c_enc    = (float*)alloc((size_t)2 * 32 * 256 * 4);
  __bf16* hbf_enc  = (__bf16*)alloc((size_t)2 * 32 * 256 * 2);
  float*  gates_e  = (float*)alloc((size_t)2 * 32 * 1024 * 4);
  __bf16* hid_bf   = (__bf16*)alloc((size_t)32 * 1024 * 2);
  __bf16* xc_bf    = (__bf16*)alloc((size_t)32 * 1024 * 2);
  __bf16* xh_bf    = (__bf16*)alloc((size_t)32 * 1024 * 2);
  float*  a_h      = (float*)alloc((size_t)32 * 512 * 4);
  float*  gates_d  = (float*)alloc((size_t)32 * 2048 * 4);
  __bf16* hall_bf  = (__bf16*)alloc((size_t)2048 * 512 * 2);

  auto cvt = [&](const float* src, __bf16* dst, int n) {
    k_f2bf<<<(n + 255) / 256, 256, 0, stream>>>(src, dst, n);
  };
  cvt(W_tr,   wtr_bf,   512 * 768);
  cvt(Wih_f,  wihf_bf,  1024 * 512);
  cvt(Wih_b,  wihb_bf,  1024 * 512);
  cvt(Whh_f,  whhf_bf,  1024 * 256);
  cvt(Whh_b,  whhb_bf,  1024 * 256);
  cvt(W_attn, wattn_bf, 512 * 1536);
  cvt(W_comb, wcomb_bf, 512 * 1024);
  cvt(Wih_d,  wihd_bf,  2048 * 512);
  cvt(Whh_d,  whhd_bf,  2048 * 512);
  cvt(W_out,  wout_bf,  32000 * 512);

  k_gather_src<<<(2048 * 768 + 255) / 256, 256, 0, stream>>>(xs, src_emb, xemb_bf);

  auto gemm = [&](const __bf16* A, int lda, const __bf16* W, int ldw,
                  const float* bias, float* oF, __bf16* oB, int ldo,
                  int M, int N, int K, int relu) {
    long jobs = (long)(M / 16) * (N / 64);
    int blocks = (int)((jobs + 7) / 8);
    k_gemm<<<blocks, 256, 0, stream>>>(A, lda, W, ldw, bias, oF, oB, ldo, M, N, K, relu);
  };

  // x = (src_emb[xs] @ W_tr^T + b_tr)  -> bf16 (2048,512)
  gemm(xemb_bf, 768, wtr_bf, 768, b_tr, nullptr, x_bf, 512, 2048, 512, 768, 0);
  // input-gate precompute (both directions) with bias folded in
  gemm(x_bf, 512, wihf_bf, 512, b_f, gi_f, nullptr, 1024, 2048, 1024, 512, 0);
  gemm(x_bf, 512, wihb_bf, 512, b_b, gi_b, nullptr, 1024, 2048, 1024, 512, 0);

  k_enc_lstm<<<2, 1024, 0, stream>>>(gi_f, gi_b, whhf_bf, whhb_bf, x_lens,
                                     out_f, out_r, dec_h, dec_c,
                                     h_enc, c_enc, hbf_enc, gates_e);

  k_assemble_enc<<<(2048 * 512 + 255) / 256, 256, 0, stream>>>(out_f, out_r, x_lens,
                                                               enc_f, enc_bf);
  // enc_proj = enc @ W_e^T   (W_e = W_attn[:, 1024:1536])
  gemm(enc_bf, 512, wattn_bf + 1024, 1536, nullptr, enc_proj, nullptr, 512,
       2048, 512, 512, 0);

  k_decoder<<<1, 1024, 0, stream>>>(ys, tgt_emb, x_lens, enc_f, enc_proj,
                                    wattn_bf, b_attn, v_attn,
                                    wcomb_bf, b_comb,
                                    wihd_bf, whhd_bf, b_d,
                                    dec_h, dec_c,
                                    hid_bf, xc_bf, xh_bf, a_h, gates_d, hall_bf);

  // logits = h_all @ W_out^T + b_out  directly into d_out (rows already (b,t) order)
  gemm(hall_bf, 512, wout_bf, 512, b_out, out, nullptr, 32000, 2048, 32000, 512, 0);

  k_logsoftmax<<<2048, 256, 0, stream>>>(out);
}
